// Attention_12859132084874
// MI455X (gfx1250) — compile-verified, matrix operand order checked by reading, not run on hardware
//
#include <hip/hip_runtime.h>
#include <cstdint>
#include <cstddef>

// ---------------- types ----------------
typedef __bf16 bf16_t;
typedef __attribute__((ext_vector_type(8)))  __bf16 bf16x8;
typedef __attribute__((ext_vector_type(16))) __bf16 bf16x16;
typedef __attribute__((ext_vector_type(8)))  float  f32x8;

// ---------------- problem constants ----------------
#define B_      4
#define N_      2048
#define DIM_    1024
#define H_      16
#define D_      64
#define INNER_  1024
#define M_      (B_*N_)      // 8192
#define BH_     (B_*H_)      // 64
#define NK_     2049         // n + 1 null token
#define NKPAD_  2080         // padded to multiple of 32
#define QTILE_  256          // q rows per block (8 waves x 32 rows)
#define CHUNK_  32
#define NCH_    (NKPAD_/CHUNK_)
// SCALE (8.0) * log2(e): folded into q so scores are already in log2 domain.
#define QSCL_   11.5415603271117087f
#define NEGBIG_ (-1e30f)

// 16-bit WMMA fragment: two contiguous 8-element (16B) chunks per lane.
// Lane L holds K = {k0..k0+7, k0+16..k0+23}, k0 = (L>=16)?8:0 (folded by caller).
__device__ __forceinline__ bf16x16 frag_ld(const bf16_t* p0, const bf16_t* p1) {
    bf16x8 lo = *(const bf16x8*)p0;
    bf16x8 hi = *(const bf16x8*)p1;
    return __builtin_shufflevector(lo, hi, 0,1,2,3,4,5,6,7,8,9,10,11,12,13,14,15);
}

__device__ __forceinline__ f32x8 wmma_bf16(bf16x16 a, bf16x16 b, f32x8 c) {
    return __builtin_amdgcn_wmma_f32_16x16x32_bf16(false, a, false, b, (short)0, c, false, false);
}

// ---------------- Tensor Data Mover (TDM) helpers ----------------
#if __has_builtin(__builtin_amdgcn_tensor_load_to_lds)
#define USE_TDM 1
typedef unsigned int u32x4 __attribute__((ext_vector_type(4)));
typedef int          i32x4 __attribute__((ext_vector_type(4)));
typedef int          i32x8 __attribute__((ext_vector_type(8)));

// 2D tile load: tile_d1 rows x tile_d0 elements (bf16, data_size=2B),
// row stride = stride0 elements, tensor dims (tdim0, tdim1) for OOB.
__device__ __forceinline__ void tdm_load_2d(void* lds_dst, const void* gsrc,
                                            unsigned tile_d0, unsigned tile_d1,
                                            unsigned long long tdim0,
                                            unsigned long long tdim1,
                                            unsigned long long stride0) {
    unsigned long long ga = (unsigned long long)gsrc;
    unsigned lds_off = (unsigned)(unsigned long long)lds_dst; // LDS aperture: low 32b = offset
    u32x4 g0;
    g0[0] = 1u;                                                // count=1 (valid user D#)
    g0[1] = lds_off;                                           // lds_addr
    g0[2] = (unsigned)(ga & 0xFFFFFFFFu);                      // global_addr[31:0]
    g0[3] = (unsigned)((ga >> 32) & 0x01FFFFFFu) | (2u << 30); // addr[56:32] | type=2
    i32x8 g1;
    g1[0] = (int)(1u << 16);                                   // data_size=1 (2B), wg_mask=0
    g1[1] = (int)((unsigned)(tdim0 & 0xFFFFu) << 16);          // tensor_dim0[15:0]
    g1[2] = (int)(((tdim0 >> 16) & 0xFFFFu) |
                  ((tdim1 & 0xFFFFu) << 16));                  // dim0[31:16] | dim1[15:0]
    g1[3] = (int)(((tdim1 >> 16) & 0xFFFFu) |
                  ((unsigned)(tile_d0 & 0xFFFFu) << 16));      // dim1[31:16] | tile_dim0
    g1[4] = (int)(tile_d1 & 0xFFFFu);                          // tile_dim1 (tile_dim2=0)
    g1[5] = (int)(stride0 & 0xFFFFFFFFu);                      // tensor_dim0_stride[31:0]
    g1[6] = (int)((stride0 >> 32) & 0xFFFFu);                  // stride[47:32]
    g1[7] = 0;
    i32x4 zz = {0, 0, 0, 0};
#if __clang_major__ >= 23
    i32x8 z8 = {0, 0, 0, 0, 0, 0, 0, 0};
    __builtin_amdgcn_tensor_load_to_lds(g0, g1, zz, zz, z8, 0);
#else
    __builtin_amdgcn_tensor_load_to_lds(g0, g1, zz, zz, 0);
#endif
}

__device__ __forceinline__ void tdm_wait() {
#if __has_builtin(__builtin_amdgcn_s_wait_tensorcnt)
    __builtin_amdgcn_s_wait_tensorcnt(0);
#else
    asm volatile("s_wait_tensorcnt 0x0" ::: "memory");
#endif
}
#else
#define USE_TDM 0
#endif

// ---------------- fp32 -> bf16 transpose-convert: d[n*K+k] = s[k*N+n] ----------------
__global__ void cvt_t_kernel(const float* __restrict__ s, bf16_t* __restrict__ d,
                             int K, int Nn) {
    int i = blockIdx.x * 256 + threadIdx.x;
    if (i < K * Nn) {
        int n = i / K, k = i % K;
        d[i] = (bf16_t)s[(size_t)k * Nn + n];
    }
}

// ---------------- LayerNorm (row = 1024) -> bf16 ----------------
__global__ __launch_bounds__(256) void ln_kernel(const float* __restrict__ x,
                                                 const float* __restrict__ g,
                                                 const float* __restrict__ be,
                                                 bf16_t* __restrict__ xn) {
    const int row  = blockIdx.x;
    const int tid  = threadIdx.x;
    const int lane = tid & 31;
    const int w    = tid >> 5;
    const float* xr = x + (size_t)row * DIM_;

    float v[4], s = 0.f, s2 = 0.f;
    #pragma unroll
    for (int i = 0; i < 4; ++i) {
        v[i] = xr[tid + i * 256];
        s += v[i]; s2 += v[i] * v[i];
    }
    #pragma unroll
    for (int off = 16; off; off >>= 1) {
        s  += __shfl_xor(s,  off, 32);
        s2 += __shfl_xor(s2, off, 32);
    }
    __shared__ float rs[8], rs2[8];
    if (lane == 0) { rs[w] = s; rs2[w] = s2; }
    __syncthreads();
    if (tid == 0) {
        float S = 0.f, S2 = 0.f;
        for (int j = 0; j < 8; ++j) { S += rs[j]; S2 += rs2[j]; }
        rs[0] = S; rs2[0] = S2;
    }
    __syncthreads();
    const float mean = rs[0] * (1.f / DIM_);
    const float var  = rs2[0] * (1.f / DIM_) - mean * mean;
    const float rstd = rsqrtf(var + 1e-5f);
    #pragma unroll
    for (int i = 0; i < 4; ++i) {
        int c = tid + i * 256;
        xn[(size_t)row * DIM_ + c] = (bf16_t)((v[i] - mean) * rstd * g[c] + be[c]);
    }
}

// ---------------- WMMA bf16 GEMM: C_f32[M_,ND] = A[M_,1024] * Bt[ND,1024]^T ----------
// Block 256 thr = 8 waves (2x4), block tile 128x256, wave tile 64x64 (4x4 WMMA
// tiles), K-step 64 (two 32-deep sub-steps per staged tile). TDM double-buffered.
template <int ND>
__global__ __launch_bounds__(256) void gemm_bf16_kernel(const bf16_t* __restrict__ A,
                                                        const bf16_t* __restrict__ Bt,
                                                        float* __restrict__ C) {
    constexpr int KD = 1024;
    __shared__ __align__(16) bf16_t as[2][128 * 64];   // [m][k]  16KB/buf
    __shared__ __align__(16) bf16_t bs[2][256 * 64];   // [n][k]  32KB/buf

    const int tid  = threadIdx.x;
    const int lane = tid & 31;
    const int w    = tid >> 5;
    const int wm   = w >> 2;            // 0..1 -> m offset wm*64
    const int wn   = w & 3;             // 0..3 -> n offset wn*64
    const int rb   = blockIdx.y * 128;
    const int cb   = blockIdx.x * 256;
    const int hi16 = lane >> 4;
    const int l15  = lane & 15;
    const int k0   = hi16 ? 8 : 0;

    f32x8 acc[4][4];
    const f32x8 z = {0.f,0.f,0.f,0.f,0.f,0.f,0.f,0.f};
    #pragma unroll
    for (int mt = 0; mt < 4; ++mt)
        #pragma unroll
        for (int nt = 0; nt < 4; ++nt) acc[mt][nt] = z;

    int buf = 0;
#if USE_TDM
    if (w == 0) {
        tdm_load_2d(&as[0][0], A  + (size_t)rb * KD, 64, 128, KD, M_, KD);
        tdm_load_2d(&bs[0][0], Bt + (size_t)cb * KD, 64, 256, KD, ND, KD);
        tdm_wait();
    }
    __syncthreads();
#endif

    for (int kk = 0; kk < KD; kk += 64) {
#if USE_TDM
        if ((kk + 64) < KD && w == 0) {    // DMA next tiles while computing
            tdm_load_2d(&as[buf ^ 1][0], A  + (size_t)rb * KD + kk + 64, 64, 128, KD, M_, KD);
            tdm_load_2d(&bs[buf ^ 1][0], Bt + (size_t)cb * KD + kk + 64, 64, 256, KD, ND, KD);
        }
#else
        __syncthreads();
        #pragma unroll
        for (int i = 0; i < 4; ++i) {      // A: 128x64
            int idx = tid + i * 256;
            int row = idx >> 3, seg = idx & 7;
            *(uint4*)(&as[0][0] + row * 64 + seg * 8) =
                *(const uint4*)(A + (size_t)(rb + row) * KD + kk + seg * 8);
        }
        #pragma unroll
        for (int i = 0; i < 8; ++i) {      // B: 256x64
            int idx = tid + i * 256;
            int row = idx >> 3, seg = idx & 7;
            *(uint4*)(&bs[0][0] + row * 64 + seg * 8) =
                *(const uint4*)(Bt + (size_t)(cb + row) * KD + kk + seg * 8);
        }
        __syncthreads();
#endif
        const bf16_t* ab = &as[buf][0];
        const bf16_t* bb = &bs[buf][0];
        #pragma unroll
        for (int kh = 0; kh < 2; ++kh) {   // two 32-deep sub-steps
            bf16x16 af[4];
            #pragma unroll
            for (int mt = 0; mt < 4; ++mt) {
                const bf16_t* p = ab + (wm * 64 + mt * 16 + l15) * 64 + kh * 32 + k0;
                af[mt] = frag_ld(p, p + 16);
            }
            #pragma unroll
            for (int nt = 0; nt < 4; ++nt) {
                const bf16_t* p = bb + (wn * 64 + nt * 16 + l15) * 64 + kh * 32 + k0;
                bf16x16 bfr = frag_ld(p, p + 16);
                #pragma unroll
                for (int mt = 0; mt < 4; ++mt)
                    acc[mt][nt] = wmma_bf16(af[mt], bfr, acc[mt][nt]);
            }
        }
#if USE_TDM
        if (w == 0) tdm_wait();
        __syncthreads();
        buf ^= 1;
#endif
    }

    // Store per C/D layout: VGPR r -> row r + hi16*8, col = lane&15 (ND constant
    // lets row strides fold into immediate offsets).
    #pragma unroll
    for (int mt = 0; mt < 4; ++mt) {
        const int rowb = rb + wm * 64 + mt * 16 + hi16 * 8;
        #pragma unroll
        for (int nt = 0; nt < 4; ++nt) {
            const int col = cb + wn * 64 + nt * 16 + l15;
            float* cp = C + (size_t)rowb * ND + col;
            #pragma unroll
            for (int r = 0; r < 8; ++r) cp[(size_t)r * ND] = acc[mt][nt][r];
        }
    }
}

// ------ L2-norm Q (folds SCALE*log2e): [M,1024] f32 -> [b*h, n, 64] bf16 ------
__global__ __launch_bounds__(256) void qnorm_kernel(const float* __restrict__ qf,
                                                    const float* __restrict__ qs,
                                                    bf16_t* __restrict__ qb) {
    const int w = threadIdx.x >> 5, lane = threadIdx.x & 31;
    const int vid = blockIdx.x * 8 + w;
    const int m = vid / H_, h = vid % H_;
    const size_t ib = (size_t)m * INNER_ + h * D_;
    float a = qf[ib + lane], c = qf[ib + lane + 32];
    float ss = a * a + c * c;
    #pragma unroll
    for (int off = 16; off; off >>= 1) ss += __shfl_xor(ss, off, 32);
    const float inv = QSCL_ / fmaxf(sqrtf(ss), 1e-12f);   // l2norm * SCALE * log2(e)
    const int b = m / N_, i = m % N_;
    const size_t o = (((size_t)(b * H_ + h)) * N_ + i) * D_;
    qb[o + lane]      = (bf16_t)(a * inv * qs[lane]);
    qb[o + lane + 32] = (bf16_t)(c * inv * qs[lane + 32]);
}

// -------- L2-norm K + V, null row 0, zero pad; V stored TRANSPOSED [bh][d][key] --------
__global__ __launch_bounds__(256) void kvnorm_kernel(const float* __restrict__ kvf,
                                                     const float* __restrict__ nullkv,
                                                     const float* __restrict__ ks,
                                                     bf16_t* __restrict__ kb,
                                                     bf16_t* __restrict__ vtg) {
    const int w = threadIdx.x >> 5, lane = threadIdx.x & 31;
    const int vid = blockIdx.x * 8 + w;
    const int bh = vid / NKPAD_, j = vid % NKPAD_;
    const int b = bh / H_, h = bh % H_;
    float ka, kc, va, vc;
    if (j == 0) {
        ka = nullkv[(0 * H_ + h) * D_ + lane];
        kc = nullkv[(0 * H_ + h) * D_ + lane + 32];
        va = nullkv[(1 * H_ + h) * D_ + lane];
        vc = nullkv[(1 * H_ + h) * D_ + lane + 32];
    } else if (j <= N_) {
        const size_t base = ((size_t)b * N_ + (j - 1)) * (2 * INNER_);
        ka = kvf[base + h * D_ + lane];
        kc = kvf[base + h * D_ + lane + 32];
        va = kvf[base + INNER_ + h * D_ + lane];
        vc = kvf[base + INNER_ + h * D_ + lane + 32];
    } else {
        ka = kc = va = vc = 0.f;
    }
    float ss = ka * ka + kc * kc;
    #pragma unroll
    for (int off = 16; off; off >>= 1) ss += __shfl_xor(ss, off, 32);
    const float inv = 1.f / fmaxf(sqrtf(ss), 1e-12f);
    const size_t ok = ((size_t)bh * NKPAD_ + j) * D_;
    kb[ok + lane]      = (bf16_t)(ka * inv * ks[lane]);
    kb[ok + lane + 32] = (bf16_t)(kc * inv * ks[lane + 32]);
    const size_t ov = (size_t)bh * D_ * NKPAD_;
    vtg[ov + (size_t)lane * NKPAD_ + j]        = (bf16_t)va;
    vtg[ov + (size_t)(lane + 32) * NKPAD_ + j] = (bf16_t)vc;
}

// ------- Flash attention: 32 q-rows/wave (2 m-tiles), 32-key chunks, TDM staging -------
// Scores arrive pre-scaled in log2 domain (q carries SCALE*log2e) -> exp2f softmax.
__global__ __launch_bounds__(256) void attn_kernel(const bf16_t* __restrict__ q,   // [BH,N,64]
                                                   const bf16_t* __restrict__ k,   // [BH,NKPAD,64]
                                                   const bf16_t* __restrict__ vtg, // [BH,64,NKPAD]
                                                   bf16_t* __restrict__ o) {       // [M,1024]
    __shared__ __align__(16) bf16_t kt[2][CHUNK_ * D_];  // [key][d]
    __shared__ __align__(16) bf16_t vt[2][D_ * CHUNK_];  // [d][key]
    __shared__ __align__(16) bf16_t plds[8][32 * CHUNK_];

    const int tid  = threadIdx.x;
    const int lane = tid & 31;
    const int w    = tid >> 5;
    const int bh   = blockIdx.x / (N_ / QTILE_);
    const int tile = blockIdx.x % (N_ / QTILE_);
    const int b    = bh / H_, h = bh % H_;
    const int hi16 = lane >> 4;
    const int l15  = lane & 15;
    const int k0   = hi16 ? 8 : 0;

    // Q fragments: 2 m-tiles x 2 K-halves
    bf16x16 qa[2][2];
    #pragma unroll
    for (int mt = 0; mt < 2; ++mt) {
        const int qrow = tile * QTILE_ + w * 32 + mt * 16 + l15;
        const bf16_t* qb = q + ((size_t)bh * N_ + qrow) * D_;
        qa[mt][0] = frag_ld(qb + k0,      qb + k0 + 16);
        qa[mt][1] = frag_ld(qb + 32 + k0, qb + 32 + k0 + 16);
    }

    const f32x8 z = {0.f,0.f,0.f,0.f,0.f,0.f,0.f,0.f};
    f32x8 acc[2][4];
    float mrun[2][8], lrun[2][8];
    #pragma unroll
    for (int mt = 0; mt < 2; ++mt) {
        #pragma unroll
        for (int dt = 0; dt < 4; ++dt) acc[mt][dt] = z;
        #pragma unroll
        for (int r = 0; r < 8; ++r) { mrun[mt][r] = NEGBIG_; lrun[mt][r] = 0.f; }
    }

    const bf16_t* kbase = k   + (size_t)bh * NKPAD_ * D_;
    const bf16_t* vbase = vtg + (size_t)bh * D_ * NKPAD_;

    int buf = 0;
#if USE_TDM
    if (w == 0) {
        tdm_load_2d(&kt[0][0], kbase, D_, CHUNK_, D_, NKPAD_, D_);
        tdm_load_2d(&vt[0][0], vbase, CHUNK_, D_, NKPAD_, D_, NKPAD_);
        tdm_wait();
    }
    __syncthreads();
#endif

    for (int c = 0; c < NCH_; ++c) {
        const int jbase = c * CHUNK_;
#if USE_TDM
        if ((c + 1) < NCH_ && w == 0) {    // DMA next chunk during compute
            tdm_load_2d(&kt[buf ^ 1][0], kbase + (size_t)(jbase + CHUNK_) * D_,
                        D_, CHUNK_, D_, NKPAD_, D_);
            tdm_load_2d(&vt[buf ^ 1][0], vbase + (jbase + CHUNK_),
                        CHUNK_, D_, NKPAD_, D_, NKPAD_);
        }
#else
        __syncthreads();
        {
            int row = tid >> 3, seg = tid & 7;          // K: 32 rows x 64
            *(uint4*)(&kt[0][0] + row * D_ + seg * 8) =
                *(const uint4*)(kbase + (size_t)(jbase + row) * D_ + seg * 8);
            int dr = tid >> 2, s2 = tid & 3;            // V: 64 rows x 32
            *(uint4*)(&vt[0][0] + dr * CHUNK_ + s2 * 8) =
                *(const uint4*)(vbase + (size_t)dr * NKPAD_ + jbase + s2 * 8);
        }
        __syncthreads();
#endif
        const bf16_t* ktb = &kt[buf][0];
        const bf16_t* vtb = &vt[buf][0];

        // S = Q * K^T -> 2 m-tiles x 2 col-tiles (K frags reused across m)
        f32x8 s[2][2] = {{z, z}, {z, z}};
        #pragma unroll
        for (int t = 0; t < 2; ++t) {
            const bf16_t* kr = ktb + (t * 16 + l15) * D_;
            #pragma unroll
            for (int kh = 0; kh < 2; ++kh) {
                bf16x16 bb = frag_ld(kr + kh * 32 + k0, kr + kh * 32 + k0 + 16);
                #pragma unroll
                for (int mt = 0; mt < 2; ++mt)
                    s[mt][t] = wmma_bf16(qa[mt][kh], bb, s[mt][t]);
            }
        }
        // Only the final chunk (keys 2048..2079) contains invalid columns.
        if (c == NCH_ - 1) {
            #pragma unroll
            for (int t = 0; t < 2; ++t) {
                const bool valid = (jbase + t * 16 + l15) < NK_;
                #pragma unroll
                for (int mt = 0; mt < 2; ++mt)
                    #pragma unroll
                    for (int r = 0; r < 8; ++r)
                        s[mt][t][r] = valid ? s[mt][t][r] : NEGBIG_;
            }
        }
        // Online softmax per row in log2 domain (xor<=8 stays within 16-lane halves)
        #pragma unroll
        for (int mt = 0; mt < 2; ++mt) {
            #pragma unroll
            for (int r = 0; r < 8; ++r) {
                float cm = fmaxf(s[mt][0][r], s[mt][1][r]);
                #pragma unroll
                for (int off = 8; off; off >>= 1) cm = fmaxf(cm, __shfl_xor(cm, off, 32));
                const float mn = fmaxf(mrun[mt][r], cm);
                const float alpha = exp2f(mrun[mt][r] - mn);
                mrun[mt][r] = mn;
                const float p0 = exp2f(s[mt][0][r] - mn);
                const float p1 = exp2f(s[mt][1][r] - mn);
                s[mt][0][r] = p0; s[mt][1][r] = p1;
                float rsum = p0 + p1;
                #pragma unroll
                for (int off = 8; off; off >>= 1) rsum += __shfl_xor(rsum, off, 32);
                lrun[mt][r] = lrun[mt][r] * alpha + rsum;
                #pragma unroll
                for (int dt = 0; dt < 4; ++dt) acc[mt][dt][r] = acc[mt][dt][r] * alpha;
            }
        }
        // Re-layout P: C-fragment -> A-fragment via wave-private LDS (32x32)
        bf16_t* pw = plds[w];
        #pragma unroll
        for (int mt = 0; mt < 2; ++mt)
            #pragma unroll
            for (int t = 0; t < 2; ++t)
                #pragma unroll
                for (int r = 0; r < 8; ++r)
                    pw[(mt * 16 + r + hi16 * 8) * CHUNK_ + t * 16 + l15] = (bf16_t)s[mt][t][r];
        bf16x16 pa[2];
        #pragma unroll
        for (int mt = 0; mt < 2; ++mt) {
            const bf16_t* pr = pw + (mt * 16 + l15) * CHUNK_ + k0;
            pa[mt] = frag_ld(pr, pr + 16);
        }
        // O += P * V (V frags reused across m)
        #pragma unroll
        for (int dt = 0; dt < 4; ++dt) {
            const bf16_t* vr = vtb + (dt * 16 + l15) * CHUNK_ + k0;
            bf16x16 vf = frag_ld(vr, vr + 16);
            #pragma unroll
            for (int mt = 0; mt < 2; ++mt)
                acc[mt][dt] = wmma_bf16(pa[mt], vf, acc[mt][dt]);
        }
#if USE_TDM
        if (w == 0) tdm_wait();
        __syncthreads();
        buf ^= 1;
#endif
    }

    #pragma unroll
    for (int mt = 0; mt < 2; ++mt)
        #pragma unroll
        for (int r = 0; r < 8; ++r) {
            const float inv = 1.f / lrun[mt][r];
            const int gm = tile * QTILE_ + w * 32 + mt * 16 + r + hi16 * 8;
            const size_t ro = ((size_t)b * N_ + gm) * INNER_ + h * D_;
            #pragma unroll
            for (int dt = 0; dt < 4; ++dt)
                o[ro + dt * 16 + l15] = (bf16_t)(acc[mt][dt][r] * inv);
        }
}

// ---------------- host launcher ----------------
extern "C" void kernel_launch(void* const* d_in, const int* in_sizes, int n_in,
                              void* d_out, int out_size, void* d_ws, size_t ws_size,
                              hipStream_t stream) {
    const float* x      = (const float*)d_in[0];
    const float* gamma  = (const float*)d_in[1];
    const float* beta   = (const float*)d_in[2];
    const float* nullkv = (const float*)d_in[3];
    const float* Wq     = (const float*)d_in[4];
    const float* Wkv    = (const float*)d_in[5];
    const float* qs     = (const float*)d_in[6];
    const float* ks     = (const float*)d_in[7];
    const float* Wo     = (const float*)d_in[8];
    float* out = (float*)d_out;

    char* ws = (char*)d_ws;
    size_t off = 0;
    auto alloc = [&](size_t bytes) { size_t o = off; off += (bytes + 255) & ~(size_t)255; return o; };

    bf16_t* xn     = (bf16_t*)(ws + alloc((size_t)M_ * DIM_ * 2));
    bf16_t* wq_t   = (bf16_t*)(ws + alloc((size_t)DIM_ * INNER_ * 2));      // [N][K]
    bf16_t* wkv_t  = (bf16_t*)(ws + alloc((size_t)DIM_ * 2 * INNER_ * 2));  // [2N][K]
    bf16_t* wo_t   = (bf16_t*)(ws + alloc((size_t)INNER_ * DIM_ * 2));      // [N][K]
    float*  q_f    = (float*) (ws + alloc((size_t)M_ * INNER_ * 4));
    float*  kv_f   = (float*) (ws + alloc((size_t)M_ * 2 * INNER_ * 4));
    bf16_t* q_b    = (bf16_t*)(ws + alloc((size_t)BH_ * N_ * D_ * 2));
    bf16_t* k_b    = (bf16_t*)(ws + alloc((size_t)BH_ * NKPAD_ * D_ * 2));
    bf16_t* vt_g   = (bf16_t*)(ws + alloc((size_t)BH_ * D_ * NKPAD_ * 2));
    bf16_t* at_b   = (bf16_t*)(ws + alloc((size_t)M_ * INNER_ * 2));

    // 1) weight transpose-converts (fp32 [K][N] -> bf16 [N][K])
    {
        int n1 = DIM_ * INNER_, n2 = DIM_ * 2 * INNER_;
        cvt_t_kernel<<<(n1 + 255) / 256, 256, 0, stream>>>(Wq,  wq_t,  DIM_, INNER_);
        cvt_t_kernel<<<(n2 + 255) / 256, 256, 0, stream>>>(Wkv, wkv_t, DIM_, 2 * INNER_);
        cvt_t_kernel<<<(n1 + 255) / 256, 256, 0, stream>>>(Wo,  wo_t,  INNER_, DIM_);
    }
    // 2) LayerNorm -> bf16
    ln_kernel<<<M_, 256, 0, stream>>>(x, gamma, beta, xn);
    // 3) Q and KV projections (WMMA + TDM)
    gemm_bf16_kernel<INNER_><<<dim3(INNER_ / 256, M_ / 128), 256, 0, stream>>>(xn, wq_t, q_f);
    gemm_bf16_kernel<2 * INNER_><<<dim3(2 * INNER_ / 256, M_ / 128), 256, 0, stream>>>(xn, wkv_t, kv_f);
    // 4) L2 norms / layout (V pre-transposed for TDM row staging)
    qnorm_kernel<<<(M_ * H_) / 8, 256, 0, stream>>>(q_f, qs, q_b);
    kvnorm_kernel<<<(BH_ * NKPAD_) / 8, 256, 0, stream>>>(kv_f, nullkv, ks, k_b, vt_g);
    // 5) flash attention (WMMA + TDM)
    attn_kernel<<<BH_ * (N_ / QTILE_), 256, 0, stream>>>(q_b, k_b, vt_g, at_b);
    // 6) output projection (WMMA + TDM) -> d_out fp32
    gemm_bf16_kernel<INNER_><<<dim3(INNER_ / 256, M_ / 128), 256, 0, stream>>>(at_b, wo_t, out);
}
